// SelfAttention3D_81063212744771
// MI455X (gfx1250) — matrix operand order repaired and checked by reading, not compiled, not run on hardware
//
#include <hip/hip_runtime.h>
#include <hip/hip_bf16.h>

// CDNA5 (gfx1250) fused flash-attention for the 3D self-attention block.
// All GEMMs on v_wmma_f32_16x16x32_f16 (wave32). Workspace holds f16
// qT (N,C), kT (N,C), V (C,N) per batch.
//
// Attention kernel: 4 waves per workgroup split the key (m) dimension
// flash-style; partials merged through LDS (max/sum rescale + accumulate).

typedef _Float16 v16h __attribute__((ext_vector_type(16)));
typedef _Float16 v8h  __attribute__((ext_vector_type(8)));
typedef float    v8f  __attribute__((ext_vector_type(8)));

#define WMMA_F16(a, b, c) \
  __builtin_amdgcn_wmma_f32_16x16x32_f16(false, (a), false, (b), (short)0, (c), false, false)

constexpr int kB = 2;
constexpr int kC = 128;
constexpr int kN = 4096;   // 16*16*16
constexpr float kScale = 1.3f;

union HF16 { v16h v; v8h h[2]; };

__device__ inline v16h pack_w16(const float* __restrict__ p) {
  v16h r;
#pragma unroll
  for (int j = 0; j < 16; ++j) r[j] = (_Float16)p[j];
  return r;
}

// ---------------------------------------------------------------------------
// Kernel 1: QKV projection. One wave computes a 16(n) x 16(o) tile of q,k,v.
//   qT[b][n][o], kT[b][n][o]  (row-major N x C, f16)
//   V [b][o][n]               (row-major C x N, f16)
// A operand = x^T tile (rows n, K = c) staged via LDS; B operand = W rows
// (contiguous f32 -> f16).
// ---------------------------------------------------------------------------
__global__ __launch_bounds__(32)
void sa3d_proj_kernel(const float* __restrict__ x,
                      const float* __restrict__ Wq, const float* __restrict__ bq,
                      const float* __restrict__ Wk, const float* __restrict__ bk,
                      const float* __restrict__ Wv, const float* __restrict__ bv,
                      _Float16* __restrict__ qT, _Float16* __restrict__ kT,
                      _Float16* __restrict__ vC) {
  __shared__ _Float16 xT[16][32];          // x^T staging: [n][c-chunk]
  const int lane = threadIdx.x;
  const int lo = lane & 15;
  const int hi = lane >> 4;                // 0 or 1
  const int hi8 = hi * 8;
  const int n0 = blockIdx.x * 16;
  const int o0 = blockIdx.y * 16;
  const int b  = blockIdx.z;

  const size_t xb = (size_t)b * kC * kN;
  v8f accq = {}, acck = {}, accv = {};

#pragma unroll
  for (int c0 = 0; c0 < kC; c0 += 32) {
    // stage x[b, c0+lane, n0..n0+15] transposed into LDS as f16
    {
      const float* xp = x + xb + (size_t)(c0 + lane) * kN + n0;
#pragma unroll
      for (int j = 0; j < 16; ++j) xT[j][lane] = (_Float16)xp[j];
    }
    __syncthreads();

    // A fragment: row n = lo, K elements 16*hi + {0..15}
    HF16 a;
    a.h[0] = *(const v8h*)&xT[lo][hi * 16];
    a.h[1] = *(const v8h*)&xT[lo][hi * 16 + 8];

    // B fragments: column o = o0+lo, K rows c0 + 16*hi + {0..15} (contig in W)
    const int wof = (o0 + lo) * kC + c0 + hi * 16;
    accq = WMMA_F16(a.v, pack_w16(Wq + wof), accq);
    acck = WMMA_F16(a.v, pack_w16(Wk + wof), acck);
    accv = WMMA_F16(a.v, pack_w16(Wv + wof), accv);
    __syncthreads();
  }

  // D layout: lane holds (n = n0 + hi8 + r, o = o0 + lo)
  const int oc = o0 + lo;
  const float biasq = bq[oc], biask = bk[oc], biasv = bv[oc];
  const size_t nb = (size_t)b * kN * kC;
#pragma unroll
  for (int r = 0; r < 8; ++r) {
    const size_t n = (size_t)(n0 + hi8 + r);
    qT[nb + n * kC + oc] = (_Float16)(accq[r] + biasq);
    kT[nb + n * kC + oc] = (_Float16)(acck[r] + biask);
  }
  v8h vpack;
#pragma unroll
  for (int r = 0; r < 8; ++r) vpack[r] = (_Float16)(accv[r] + biasv);
  *(v8h*)(vC + (size_t)b * kC * kN + (size_t)oc * kN + n0 + hi8) = vpack;
}

// ---------------------------------------------------------------------------
// Kernel 2: fused flash attention + residual. One workgroup (4 waves) owns a
// 16-query tile; each wave processes an interleaved quarter of the keys.
// S^T tiles (M=m, N=n) keep softmax state per-lane and let P repack in-lane
// into the A-operand layout of O^T += P * V^T (f32 accumulation).
// Partials merged flash-style through LDS; deterministic (no atomics).
// ---------------------------------------------------------------------------
__global__ __launch_bounds__(128)
void sa3d_attn_kernel(const float* __restrict__ x, const float* __restrict__ gamma_p,
                      const _Float16* __restrict__ qT, const _Float16* __restrict__ kT,
                      const _Float16* __restrict__ vC, float* __restrict__ out) {
  __shared__ float smax_s[4][16];
  __shared__ float ssum_s[4][16];
  __shared__ float opart[4][8][32][8];   // [wave][ctile][lane][r] = 32 KB

  const int tid  = threadIdx.x;
  const int lane = tid & 31;
  const int w    = tid >> 5;             // wave id within workgroup, 0..3
  const int lo = lane & 15;
  const int hi = lane >> 4;
  const int hi8 = hi * 8;
  const int n0 = blockIdx.x * 16;
  const int b  = blockIdx.y;
  const float gamma = *gamma_p;

  const _Float16* qTb = qT + (size_t)b * kN * kC;
  const _Float16* kTb = kT + (size_t)b * kN * kC;
  const _Float16* vb  = vC + (size_t)b * kC * kN;

  // Preload B_q fragments: column n = n0+lo, K rows c-chunk + {hi8..hi8+7} and {hi8+16..hi8+23}
  HF16 bqf[4];
#pragma unroll
  for (int ck = 0; ck < 4; ++ck) {
    const _Float16* p = qTb + (size_t)(n0 + lo) * kC + ck * 32 + hi8;
    bqf[ck].h[0] = *(const v8h*)p;
    bqf[ck].h[1] = *(const v8h*)(p + 16);
  }

  v8f zero = {};
  v8f oacc[8];
#pragma unroll
  for (int t = 0; t < 8; ++t) oacc[t] = zero;

  float mrun = -1.0e30f;   // running max of scaled logits (per lane's n)
  float lrun = 0.0f;       // running sum of exp

  // Each wave handles keys m0 = w*32, w*32+128, ... (32 steps)
  for (int m0 = w * 32; m0 < kN; m0 += 128) {
    // ---- S^T tiles for key rows [m0, m0+16) and [m0+16, m0+32) ----
    v8f s0 = zero, s1 = zero;
#pragma unroll
    for (int ck = 0; ck < 4; ++ck) {
      HF16 a0, a1;
      const _Float16* p0 = kTb + (size_t)(m0 + lo) * kC + ck * 32 + hi8;
      a0.h[0] = *(const v8h*)p0;
      a0.h[1] = *(const v8h*)(p0 + 16);
      const _Float16* p1 = kTb + (size_t)(m0 + 16 + lo) * kC + ck * 32 + hi8;
      a1.h[0] = *(const v8h*)p1;
      a1.h[1] = *(const v8h*)(p1 + 16);
      s0 = WMMA_F16(a0.v, bqf[ck].v, s0);
      s1 = WMMA_F16(a1.v, bqf[ck].v, s1);
    }

    // ---- online softmax update (lane's query n = n0+lo; partner lane ^16) ----
    float sv0[8], sv1[8];
    float mloc = -1.0e30f;
#pragma unroll
    for (int r = 0; r < 8; ++r) {
      sv0[r] = kScale * s0[r];
      sv1[r] = kScale * s1[r];
      mloc = fmaxf(mloc, fmaxf(sv0[r], sv1[r]));
    }
    mloc = fmaxf(mloc, __shfl_xor(mloc, 16, 32));
    const float mnew = fmaxf(mrun, mloc);
    const float corr = __expf(mrun - mnew);

    HF16 pa;                 // P in A-operand layout, no cross-lane movement
    float lsum = 0.0f;
#pragma unroll
    for (int r = 0; r < 8; ++r) {
      const float e0 = __expf(sv0[r] - mnew);
      const float e1 = __expf(sv1[r] - mnew);
      lsum += e0 + e1;
      pa.v[r]     = (_Float16)e0;   // K = hi8 + r
      pa.v[r + 8] = (_Float16)e1;   // K = hi8 + 16 + r
    }
    lsum += __shfl_xor(lsum, 16, 32);
    lrun = lrun * corr + lsum;
    mrun = mnew;

    // broadcast per-row correction into O^T slot order (row n = hi8 + r)
    float cs[8];
#pragma unroll
    for (int r = 0; r < 8; ++r) cs[r] = __shfl(corr, hi8 + r, 32);
#pragma unroll
    for (int t = 0; t < 8; ++t)
#pragma unroll
      for (int r = 0; r < 8; ++r) oacc[t][r] *= cs[r];

    // ---- O^T += P * V^T over the 8 c-tiles ----
#pragma unroll
    for (int t = 0; t < 8; ++t) {
      HF16 bvf;  // column c = t*16+lo, K rows m0 + {hi8..hi8+7} and {hi8+16..hi8+23}
      const _Float16* p = vb + (size_t)(t * 16 + lo) * kN + m0 + hi8;
      bvf.h[0] = *(const v8h*)p;
      bvf.h[1] = *(const v8h*)(p + 16);
      oacc[t] = WMMA_F16(pa.v, bvf.v, oacc[t]);
    }
  }

  // ---- flash merge across the 4 waves via LDS ----
  if (lane < 16) {
    smax_s[w][lane] = mrun;   // lane's query n = lane
    ssum_s[w][lane] = lrun;
  }
  __syncthreads();

  float Mr[8], fr[8];
#pragma unroll
  for (int r = 0; r < 8; ++r) {
    const int nr = hi8 + r;   // row n of accumulator slot r
    const float M = fmaxf(fmaxf(smax_s[0][nr], smax_s[1][nr]),
                          fmaxf(smax_s[2][nr], smax_s[3][nr]));
    Mr[r] = M;
    fr[r] = __expf(smax_s[w][nr] - M);
  }
#pragma unroll
  for (int t = 0; t < 8; ++t)
#pragma unroll
    for (int r = 0; r < 8; ++r) opart[w][t][lane][r] = oacc[t][r] * fr[r];
  __syncthreads();

  // global normalizer per accumulator slot
  float invL[8];
#pragma unroll
  for (int r = 0; r < 8; ++r) {
    const int nr = hi8 + r;
    float L = 0.0f;
#pragma unroll
    for (int w2 = 0; w2 < 4; ++w2)
      L += ssum_s[w2][nr] * __expf(smax_s[w2][nr] - Mr[r]);
    invL[r] = 1.0f / L;
  }

  // each wave finalizes 2 c-tiles: lane holds (n = n0+hi8+r, c = t*16+lo)
  const size_t xb = (size_t)b * kC * kN;
#pragma unroll
  for (int tt = 0; tt < 2; ++tt) {
    const int t = w * 2 + tt;
    float os[8];
#pragma unroll
    for (int r = 0; r < 8; ++r)
      os[r] = opart[0][t][lane][r] + opart[1][t][lane][r] +
              opart[2][t][lane][r] + opart[3][t][lane][r];

    const int c = t * 16 + lo;
    const float* xp = x + xb + (size_t)c * kN + n0 + hi8;
    float* op = out + xb + (size_t)c * kN + n0 + hi8;
    const float4 x0 = *(const float4*)xp;
    const float4 x1 = *(const float4*)(xp + 4);
    float4 r0, r1;
    r0.x = x0.x + gamma * os[0] * invL[0];
    r0.y = x0.y + gamma * os[1] * invL[1];
    r0.z = x0.z + gamma * os[2] * invL[2];
    r0.w = x0.w + gamma * os[3] * invL[3];
    r1.x = x1.x + gamma * os[4] * invL[4];
    r1.y = x1.y + gamma * os[5] * invL[5];
    r1.z = x1.z + gamma * os[6] * invL[6];
    r1.w = x1.w + gamma * os[7] * invL[7];
    *(float4*)op = r0;
    *(float4*)(op + 4) = r1;
  }
}

// ---------------------------------------------------------------------------
extern "C" void kernel_launch(void* const* d_in, const int* in_sizes, int n_in,
                              void* d_out, int out_size, void* d_ws, size_t ws_size,
                              hipStream_t stream) {
  const float* x     = (const float*)d_in[0];
  const float* Wq    = (const float*)d_in[1];
  const float* bq    = (const float*)d_in[2];
  const float* Wk    = (const float*)d_in[3];
  const float* bk    = (const float*)d_in[4];
  const float* Wv    = (const float*)d_in[5];
  const float* bv    = (const float*)d_in[6];
  const float* gamma = (const float*)d_in[7];
  float* out = (float*)d_out;

  // workspace: qT, kT as (B,N,C) f16; V as (B,C,N) f16  -> 6 MB total
  _Float16* qT = (_Float16*)d_ws;
  _Float16* kT = qT + (size_t)kB * kN * kC;
  _Float16* vC = kT + (size_t)kB * kN * kC;

  dim3 g1(kN / 16, kC / 16, kB);     // 4096 waves
  sa3d_proj_kernel<<<g1, 32, 0, stream>>>(x, Wq, bq, Wk, bk, Wv, bv, qT, kT, vC);

  dim3 g2(kN / 16, kB);              // 512 blocks x 4 waves = 2048 waves
  sa3d_attn_kernel<<<g2, 128, 0, stream>>>(x, gamma, qT, kT, vC, out);
}
